// Encoder_18365280158054
// MI455X (gfx1250) — compile-verified, matrix operand order checked by reading, not run on hardware
//
#include <hip/hip_runtime.h>
#include <hip/hip_bf16.h>
#include <math.h>
#include <stdint.h>

typedef __attribute__((ext_vector_type(16))) _Float16 v16h;
typedef __attribute__((ext_vector_type(8)))  float    v8f;

#define LAYERS 12
#define HDIM   768
#define H2DIM  1536
#define NHEAD  12
#define DHEAD  64
#define FFI    2048
#define SEQ    512
#define BSZ    4
#define SB     2048      /* SEQ*BSZ */
#define KPOS   63
#define EPSV   1e-7f

// ---------------------------------------------------------------------------
// Generic strided batched GEMM:  C[b,h,m,n] = sum_k A[b,h,m,k] * B[b,h,n,k]
//                                (+ bias[n]) (+ res[b,h,m,n])
// A/B are f16 in global memory; C written fp32 and/or f16.
// Block: 256 threads (8 waves), tile 128(M) x 128(N), K-step 32.
// Waves 4(M) x 2(N): each wave owns 32x64 -> 2 A-frags, 4 B-frags, 8 WMMAs/step.
// Staging: per-lane GLOBAL_LOAD_ASYNC_TO_LDS_B128 (ASYNCcnt) when k-stride==1.
// ---------------------------------------------------------------------------
struct GemmDesc {
  const _Float16* A; const _Float16* B;
  float* C; _Float16* C16;
  const float* bias; const float* res;
  int M, N, K, nH;
  long aB, aH, aM, aK;
  long bB, bH, bN, bK;
  long cB, cH, cM, cN;
};

// Copy 32 contiguous bytes global -> LDS without touching VGPR data paths.
__device__ inline void async_copy32B(_Float16* dst, const _Float16* src) {
  unsigned           lds = (unsigned)(uintptr_t)dst;            // LDS byte offset
  unsigned long long ga  = (unsigned long long)(uintptr_t)src;  // global address
  asm volatile(
      "global_load_async_to_lds_b128 %0, %1, off\n\t"
      "global_load_async_to_lds_b128 %0, %1, off offset:16"
      :: "v"(lds), "v"(ga) : "memory");
}

template<bool VEC>
__device__ inline void stage_tile(_Float16 (*S)[40], const _Float16* base,
                                  int grow, int limit, long rstride, long kstride,
                                  int k0, int sk, int sr) {
  _Float16* dst = &S[sr][sk];
  if (grow < limit) {
    const _Float16* src = base + (long)grow * rstride + (long)(k0 + sk) * kstride;
    if (VEC) {
      async_copy32B(dst, src);      // 16 halfs, ASYNCcnt-tracked
    } else {
      #pragma unroll
      for (int i = 0; i < 16; ++i) dst[i] = src[(long)i * kstride];
    }
  } else {
    #pragma unroll
    for (int i = 0; i < 16; ++i) dst[i] = (_Float16)0.0f;
  }
}

template<bool AVEC, bool BVEC>
__global__ __launch_bounds__(256) void gemm_wmma_kernel(GemmDesc g) {
  const int bi = blockIdx.z;
  const int b  = bi / g.nH;
  const int h  = bi % g.nH;
  const _Float16* Ab = g.A + (long)b * g.aB + (long)h * g.aH;
  const _Float16* Bb = g.B + (long)b * g.bB + (long)h * g.bH;
  float*     Cb  = g.C   ? (g.C   + (long)b * g.cB + (long)h * g.cH) : nullptr;
  _Float16*  C16 = g.C16 ? (g.C16 + (long)b * g.cB + (long)h * g.cH) : nullptr;
  const float* Rb = g.res ? (g.res + (long)b * g.cB + (long)h * g.cH) : nullptr;

  __shared__ _Float16 As[128][40];   // rows padded to 80B: 16B-aligned staging
  __shared__ _Float16 Bs[128][40];

  const int tid  = threadIdx.x;
  const int lane = tid & 31;
  const int wave = tid >> 5;
  const int wm   = wave & 3;        // 0..3 -> 32-row strip
  const int wn   = wave >> 2;       // 0..1 -> 64-col strip
  const int m0   = blockIdx.x * 128;
  const int n0   = blockIdx.y * 128;

  const int sr = tid >> 1;          // staging row 0..127
  const int sk = (tid & 1) * 16;    // staging k 0 or 16

  v8f acc[2][4];
  #pragma unroll
  for (int i = 0; i < 2; ++i)
    #pragma unroll
    for (int j = 0; j < 4; ++j) acc[i][j] = (v8f){};

  for (int k0 = 0; k0 < g.K; k0 += 32) {
    stage_tile<AVEC>(As, Ab, m0 + sr, g.M, g.aM, g.aK, k0, sk, sr);
    stage_tile<BVEC>(Bs, Bb, n0 + sr, g.N, g.bN, g.bK, k0, sk, sr);
    if (AVEC || BVEC)
      asm volatile("s_wait_asynccnt 0x0" ::: "memory");
    __syncthreads();

    // fragments per CDNA5 16-bit A layout (ISA 7.12.2)
    const int lm   = lane & 15;
    const int koff = (lane < 16) ? 0 : 8;
    v16h af[2], bf[4];
    #pragma unroll
    for (int r = 0; r < 8; ++r) {
      const int kb = ((r & 4) << 2) + koff + 2 * (r & 3);   // K index of half pair
      #pragma unroll
      for (int i = 0; i < 2; ++i) {
        af[i][2*r]   = As[wm * 32 + i * 16 + lm][kb];
        af[i][2*r+1] = As[wm * 32 + i * 16 + lm][kb + 1];
      }
      #pragma unroll
      for (int j = 0; j < 4; ++j) {
        bf[j][2*r]   = Bs[wn * 64 + j * 16 + lm][kb];
        bf[j][2*r+1] = Bs[wn * 64 + j * 16 + lm][kb + 1];
      }
    }
    #pragma unroll
    for (int i = 0; i < 2; ++i)
      #pragma unroll
      for (int j = 0; j < 4; ++j)
        acc[i][j] = __builtin_amdgcn_wmma_f32_16x16x32_f16(
            false, af[i], false, bf[j], (short)0, acc[i][j], false, false);
    __syncthreads();
  }

  // epilogue: C layout (ISA 7.12.2 32-bit C/D 16x16)
  const int lm = lane & 15;
  const int rb = (lane < 16) ? 0 : 8;
  #pragma unroll
  for (int i = 0; i < 2; ++i) {
    #pragma unroll
    for (int r = 0; r < 8; ++r) {
      const int gm = m0 + wm * 32 + i * 16 + rb + r;
      if (gm >= g.M) continue;
      #pragma unroll
      for (int j = 0; j < 4; ++j) {
        const int gn = n0 + wn * 64 + j * 16 + lm;
        if (gn < g.N) {
          float v = acc[i][j][r];
          if (g.bias) v += g.bias[gn];
          const long ci = (long)gm * g.cM + (long)gn * g.cN;
          if (Rb)  v += Rb[ci];
          if (Cb)  Cb[ci] = v;
          if (C16) C16[ci] = (_Float16)v;
        }
      }
    }
  }
}

// ---------------------------------------------------------------------------
// Block reductions (256 threads)
// ---------------------------------------------------------------------------
__device__ inline float block_sum(float v, float* sbuf) {
  const int t = threadIdx.x;
  sbuf[t] = v; __syncthreads();
  for (int s = 128; s > 0; s >>= 1) {
    if (t < s) sbuf[t] += sbuf[t + s];
    __syncthreads();
  }
  float r = sbuf[0];
  __syncthreads();
  return r;
}

// ---------------------------------------------------------------------------
// fp32 -> f16 conversion
// ---------------------------------------------------------------------------
__global__ __launch_bounds__(256) void cvt_f16_kernel(const float* in, _Float16* out, long n) {
  const long i = (long)blockIdx.x * 256 + threadIdx.x;
  if (i < n) out[i] = (_Float16)in[i];
}

// ---------------------------------------------------------------------------
// Plain LayerNorm (no affine), f16 output (feeds WMMA GEMMs only); cols = 768
// ---------------------------------------------------------------------------
__global__ __launch_bounds__(256) void ln_kernel(const float* in, _Float16* out, int cols) {
  __shared__ float sbuf[256];
  const long row = blockIdx.x;
  const float* x = in + row * cols;
  _Float16* o = out + row * cols;
  const int per = cols >> 8;
  float v[8];
  float s = 0.0f;
  for (int i = 0; i < per; ++i) {
    v[i] = x[threadIdx.x + (i << 8)];
    s += v[i];
  }
  const float mean = block_sum(s, sbuf) / (float)cols;
  float ss = 0.0f;
  for (int i = 0; i < per; ++i) { float d = v[i] - mean; ss += d * d; }
  const float inv = rsqrtf(block_sum(ss, sbuf) / (float)cols + EPSV);
  for (int i = 0; i < per; ++i)
    o[threadIdx.x + (i << 8)] = (_Float16)((v[i] - mean) * inv);
}

// ---------------------------------------------------------------------------
// x_mid = x_in + ln(attn)*gamma + beta          (cols = 768, fp32 out)
// ---------------------------------------------------------------------------
__global__ __launch_bounds__(256) void ln_affres_kernel(const float* xin, const float* attn,
                                                        const float* gamma, const float* beta,
                                                        float* out) {
  __shared__ float sbuf[256];
  const long row = blockIdx.x;
  const float* a = attn + row * HDIM;
  const float* xi = xin + row * HDIM;
  float* o = out + row * HDIM;
  float v[3];
  float s = 0.0f;
  for (int i = 0; i < 3; ++i) { v[i] = a[threadIdx.x + (i << 8)]; s += v[i]; }
  const float mean = block_sum(s, sbuf) / (float)HDIM;
  float ss = 0.0f;
  for (int i = 0; i < 3; ++i) { float d = v[i] - mean; ss += d * d; }
  const float inv = rsqrtf(block_sum(ss, sbuf) / (float)HDIM + EPSV);
  for (int i = 0; i < 3; ++i) {
    const int c = threadIdx.x + (i << 8);
    o[c] = xi[c] + (v[i] - mean) * inv * gamma[c] + beta[c];
  }
}

// ---------------------------------------------------------------------------
// GEGLU + LayerNorm: act = a * gelu_new(gate), out = f16(ln(act))  (cols=2048)
// ---------------------------------------------------------------------------
__global__ __launch_bounds__(256) void geglu_ln_kernel(const float* u, _Float16* out) {
  __shared__ float sbuf[256];
  const long row = blockIdx.x;
  const float* a  = u + row * (2 * FFI);
  const float* gt = a + FFI;
  _Float16* o = out + row * FFI;
  float v[8];
  float s = 0.0f;
  for (int i = 0; i < 8; ++i) {
    const int c = threadIdx.x + (i << 8);
    const float x = gt[c];
    const float t = 0.5f * x * (1.0f + tanhf(0.7978845608028654f * (x + 0.044715f * x * x * x)));
    v[i] = a[c] * t;
    s += v[i];
  }
  const float mean = block_sum(s, sbuf) / (float)FFI;
  float ss = 0.0f;
  for (int i = 0; i < 8; ++i) { float d = v[i] - mean; ss += d * d; }
  const float inv = rsqrtf(block_sum(ss, sbuf) / (float)FFI + EPSV);
  for (int i = 0; i < 8; ++i)
    o[threadIdx.x + (i << 8)] = (_Float16)((v[i] - mean) * inv);
}

// ---------------------------------------------------------------------------
// Assemble scores + relative-position gathers + mask + softmax -> probs
// Writes fp32 probs (model output) and f16 probs (feeds probs@V WMMA GEMM).
// block per (bh, q): grid (SEQ, BSZ*NHEAD), 256 threads, 2 keys/thread
// ---------------------------------------------------------------------------
__global__ __launch_bounds__(256) void attn_softmax_kernel(const float* sraw, const float* cp,
                                                           const float* pcT, const int* idx,
                                                           const unsigned char* mask,
                                                           float* probs, _Float16* probs16,
                                                           float scale) {
  __shared__ float sbuf[256];
  const int q  = blockIdx.x;
  const int bh = blockIdx.y;
  const int b  = bh / NHEAD;
  const float* srow = sraw + ((long)bh * SEQ + q) * SEQ;
  const float* cpr  = cp   + ((long)bh * SEQ + q) * 64;
  const float* pcb  = pcT  + (long)bh * SEQ * 64;
  const int*   ir   = idx  + (long)q * SEQ;
  const unsigned char* mr = mask + (long)b * SEQ;
  float*    orow  = probs   + ((long)bh * SEQ + q) * SEQ;
  _Float16* orow2 = probs16 + ((long)bh * SEQ + q) * SEQ;

  const float NEG_INF = -__builtin_inff();
  float vals[2];
  bool  msk[2];
  float mx = NEG_INF;
  #pragma unroll
  for (int j = 0; j < 2; ++j) {
    const int k = threadIdx.x + (j << 8);
    const int id = ir[k];
    float v = scale * (srow[k] + cpr[id] + pcb[(long)k * 64 + id]);
    msk[j] = (mr[k] != 0);
    if (msk[j]) v = NEG_INF;
    vals[j] = v;
    mx = fmaxf(mx, v);
  }
  sbuf[threadIdx.x] = mx; __syncthreads();
  for (int s = 128; s > 0; s >>= 1) {
    if (threadIdx.x < s) sbuf[threadIdx.x] = fmaxf(sbuf[threadIdx.x], sbuf[threadIdx.x + s]);
    __syncthreads();
  }
  mx = sbuf[0]; __syncthreads();

  float sum = 0.0f;
  #pragma unroll
  for (int j = 0; j < 2; ++j) {
    vals[j] = msk[j] ? 0.0f : expf(vals[j] - mx);
    sum += vals[j];
  }
  const float tot = block_sum(sum, sbuf);
  const float inv = 1.0f / tot;
  #pragma unroll
  for (int j = 0; j < 2; ++j) {
    const int k = threadIdx.x + (j << 8);
    const float p = vals[j] * inv;
    orow[k]  = p;
    orow2[k] = (_Float16)p;
  }
}

__global__ __launch_bounds__(256) void copy_kernel(const float* in, float* out) {
  const long i = (long)blockIdx.x * 256 + threadIdx.x;
  out[i] = in[i];
}

// ---------------------------------------------------------------------------
static inline void launch_gemm(hipStream_t st, const _Float16* A, const _Float16* B,
                               float* C, _Float16* C16,
                               const float* bias, const float* res,
                               int M, int N, int K, int nBatch, int nH,
                               long aB, long aH, long aM, long aK,
                               long bB, long bH, long bN, long bK,
                               long cB, long cH, long cM, long cN) {
  GemmDesc g;
  g.A = A; g.B = B; g.C = C; g.C16 = C16; g.bias = bias; g.res = res;
  g.M = M; g.N = N; g.K = K; g.nH = nH;
  g.aB = aB; g.aH = aH; g.aM = aM; g.aK = aK;
  g.bB = bB; g.bH = bH; g.bN = bN; g.bK = bK;
  g.cB = cB; g.cH = cH; g.cM = cM; g.cN = cN;
  dim3 grid((M + 127) / 128, (N + 127) / 128, nBatch);
  dim3 blk(256);
  if (bK == 1) gemm_wmma_kernel<true, true ><<<grid, blk, 0, st>>>(g);
  else         gemm_wmma_kernel<true, false><<<grid, blk, 0, st>>>(g);
}

static inline void cvt(hipStream_t st, const float* in, _Float16* out, long n) {
  cvt_f16_kernel<<<(int)((n + 255) / 256), 256, 0, st>>>(in, out, n);
}

extern "C" void kernel_launch(void* const* d_in, const int* in_sizes, int n_in,
                              void* d_out, int out_size, void* d_ws, size_t ws_size,
                              hipStream_t stream) {
  (void)in_sizes; (void)n_in; (void)out_size; (void)ws_size;

  const float*         hidden = (const float*)d_in[0];
  const unsigned char* mask   = (const unsigned char*)d_in[1];   // bool [B,1,1,S]
  const float*         rel    = (const float*)d_in[2];           // [63, H]
  const int*           pidx   = (const int*)d_in[3];             // [512, 512]
  const float*         Wqk    = (const float*)d_in[4];
  const float*         bqk    = (const float*)d_in[5];
  const float*         Wv     = (const float*)d_in[6];
  const float*         bv     = (const float*)d_in[7];
  const float*         Wo     = (const float*)d_in[8];
  const float*         bo     = (const float*)d_in[9];
  const float*         lng    = (const float*)d_in[10];
  const float*         lnb    = (const float*)d_in[11];
  const float*         W1     = (const float*)d_in[12];
  const float*         W2     = (const float*)d_in[13];
  float* out = (float*)d_out;

  // ---- workspace carve-up (256B-aligned) ----
  char* wsp = (char*)d_ws;
  auto alloc = [&](size_t bytes) -> char* {
    char* r = wsp;
    wsp += (bytes + 255) & ~(size_t)255;
    return r;
  };
  const long SLICE = (long)SB * HDIM;                      // 1,572,864 elems
  float*    sraw   = (float*)   alloc((size_t)BSZ*NHEAD*SEQ*SEQ * 4);
  float*    cp     = (float*)   alloc((size_t)BSZ*NHEAD*SEQ*64 * 4);
  float*    pcT    = (float*)   alloc((size_t)BSZ*NHEAD*SEQ*64 * 4);
  float*    attn   = (float*)   alloc((size_t)SLICE * 4);
  float*    xmid   = (float*)   alloc((size_t)SLICE * 4);
  float*    ubuf   = (float*)   alloc((size_t)SB*2*FFI * 4);
  _Float16* h16    = (_Float16*)alloc((size_t)SLICE * 2);
  _Float16* qk16   = (_Float16*)alloc((size_t)SB*H2DIM * 2);
  _Float16* v16    = (_Float16*)alloc((size_t)SLICE * 2);
  _Float16* pos16  = (_Float16*)alloc((size_t)64*H2DIM * 2);
  _Float16* ctx16  = (_Float16*)alloc((size_t)SLICE * 2);
  _Float16* actl16 = (_Float16*)alloc((size_t)SB*FFI * 2);
  _Float16* pr16   = (_Float16*)alloc((size_t)BSZ*NHEAD*SEQ*SEQ * 2);
  _Float16* rel16  = (_Float16*)alloc((size_t)KPOS*HDIM * 2);
  _Float16* wqk16  = (_Float16*)alloc((size_t)H2DIM*HDIM * 2);
  _Float16* wv16   = (_Float16*)alloc((size_t)HDIM*HDIM * 2);
  _Float16* wo16   = (_Float16*)alloc((size_t)HDIM*HDIM * 2);
  _Float16* w116   = (_Float16*)alloc((size_t)2*FFI*HDIM * 2);
  _Float16* w216   = (_Float16*)alloc((size_t)HDIM*FFI * 2);

  float* probs_base = out + (long)(LAYERS + 1) * SLICE;
  const float scale = 1.0f / sqrtf(3.0f * (float)DHEAD);
  const long PROBS_L = (long)BSZ * NHEAD * SEQ * SEQ;

  // hidden_stack[0] = input; rel embedding to f16 once
  copy_kernel<<<(int)(SLICE / 256), 256, 0, stream>>>(hidden, out);
  cvt(stream, rel, rel16, (long)KPOS * HDIM);

  for (int l = 0; l < LAYERS; ++l) {
    const float* x      = out + (long)l * SLICE;
    float*       xnext  = out + (long)(l + 1) * SLICE;
    float*       probsl = probs_base + (long)l * PROBS_L;

    // per-layer weight slices -> f16
    cvt(stream, Wqk + (long)l * H2DIM * HDIM, wqk16, (long)H2DIM * HDIM);
    cvt(stream, Wv  + (long)l * HDIM * HDIM,  wv16,  (long)HDIM * HDIM);
    cvt(stream, Wo  + (long)l * HDIM * HDIM,  wo16,  (long)HDIM * HDIM);
    cvt(stream, W1  + (long)l * 2 * FFI * HDIM, w116, (long)2 * FFI * HDIM);
    cvt(stream, W2  + (long)l * HDIM * FFI,   w216,  (long)HDIM * FFI);

    // h = ln(x)  (f16)
    ln_kernel<<<SB, 256, 0, stream>>>(x, h16, HDIM);

    // qk = h @ Wqk^T + bqk   [2048,1536] f16
    launch_gemm(stream, h16, wqk16, nullptr, qk16,
                bqk + (long)l * H2DIM, nullptr,
                SB, H2DIM, HDIM, 1, 1,
                0, 0, HDIM, 1,   0, 0, HDIM, 1,   0, 0, H2DIM, 1);
    // v = h @ Wv^T + bv      [2048,768] f16
    launch_gemm(stream, h16, wv16, nullptr, v16,
                bv + (long)l * HDIM, nullptr,
                SB, HDIM, HDIM, 1, 1,
                0, 0, HDIM, 1,   0, 0, HDIM, 1,   0, 0, HDIM, 1);
    // pos = rel @ Wqk^T + bqk  [63,1536] f16
    launch_gemm(stream, rel16, wqk16, nullptr, pos16,
                bqk + (long)l * H2DIM, nullptr,
                KPOS, H2DIM, HDIM, 1, 1,
                0, 0, HDIM, 1,   0, 0, HDIM, 1,   0, 0, H2DIM, 1);

    // scores_raw[b,h,q,k] = q . k   (batched over 48 heads, K=64)
    launch_gemm(stream, qk16, qk16 + HDIM, sraw, nullptr, nullptr, nullptr,
                SEQ, SEQ, DHEAD, BSZ * NHEAD, NHEAD,
                H2DIM, 64, (long)BSZ * H2DIM, 1,
                H2DIM, 64, (long)BSZ * H2DIM, 1,
                (long)NHEAD * SEQ * SEQ, (long)SEQ * SEQ, SEQ, 1);
    // c_p[b,h,q,r] = q . kpos[r]    (N=63, stride 64)
    launch_gemm(stream, qk16, pos16 + HDIM, cp, nullptr, nullptr, nullptr,
                SEQ, KPOS, DHEAD, BSZ * NHEAD, NHEAD,
                H2DIM, 64, (long)BSZ * H2DIM, 1,
                0, 64, H2DIM, 1,
                (long)NHEAD * SEQ * 64, (long)SEQ * 64, 64, 1);
    // pcT[b,h,k,r] = k . qpos[r]
    launch_gemm(stream, qk16 + HDIM, pos16, pcT, nullptr, nullptr, nullptr,
                SEQ, KPOS, DHEAD, BSZ * NHEAD, NHEAD,
                H2DIM, 64, (long)BSZ * H2DIM, 1,
                0, 64, H2DIM, 1,
                (long)NHEAD * SEQ * 64, (long)SEQ * 64, 64, 1);

    // gathers + mask + softmax -> probs (fp32 to d_out, f16 for next GEMM)
    attn_softmax_kernel<<<dim3(SEQ, BSZ * NHEAD), 256, 0, stream>>>(
        sraw, cp, pcT, pidx, mask, probsl, pr16, scale);

    // ctx[s,b,h*64+d] = sum_k probs[b,h,s,k] * v[b,h,k,d]  (fused head transpose)
    // B (v) has strided k -> scalar-gather staging path
    launch_gemm(stream, pr16, v16, nullptr, ctx16, nullptr, nullptr,
                SEQ, DHEAD, SEQ, BSZ * NHEAD, NHEAD,
                (long)NHEAD * SEQ * SEQ, (long)SEQ * SEQ, SEQ, 1,
                HDIM, 64, 1, (long)BSZ * HDIM,
                HDIM, 64, (long)BSZ * HDIM, 1);

    // attn = ctx @ Wo^T + bo  (fp32, feeds LN)
    launch_gemm(stream, ctx16, wo16, attn, nullptr,
                bo + (long)l * HDIM, nullptr,
                SB, HDIM, HDIM, 1, 1,
                0, 0, HDIM, 1,   0, 0, HDIM, 1,   0, 0, HDIM, 1);

    // xmid = x + ln(attn)*g + b
    ln_affres_kernel<<<SB, 256, 0, stream>>>(x, attn,
                                             lng + (long)l * HDIM,
                                             lnb + (long)l * HDIM, xmid);
    // hh = ln(xmid)  (f16)
    ln_kernel<<<SB, 256, 0, stream>>>(xmid, h16, HDIM);

    // u = hh @ W1^T   [2048,4096] fp32 (feeds GEGLU)
    launch_gemm(stream, h16, w116, ubuf, nullptr, nullptr, nullptr,
                SB, 2 * FFI, HDIM, 1, 1,
                0, 0, HDIM, 1,   0, 0, HDIM, 1,   0, 0, (long)2 * FFI, 1);

    // actl = f16(ln(a * gelu_new(gate)))
    geglu_ln_kernel<<<SB, 256, 0, stream>>>(ubuf, actl16);

    // xnext = xmid + actl @ W2^T   (residual fused in GEMM epilogue)
    launch_gemm(stream, actl16, w216, xnext, nullptr, nullptr, xmid,
                SB, HDIM, FFI, 1, 1,
                0, 0, FFI, 1,   0, 0, FFI, 1,   0, 0, HDIM, 1);
  }
}
